// GRUBaseline_64037962383364
// MI455X (gfx1250) — matrix operand order repaired and checked by reading, compiled, not run on hardware
//
#include <hip/hip_runtime.h>
#include <hip/hip_bf16.h>

// GRU: B=64, T=2048, I=256, H=512 (3H=1536)
// 32 persistent WGs, each owns a 16-wide slice of H:
//   - 48 rows of W_hh (48x512) + 48 rows of W_ih (48x256) resident in LDS (bf16, 72KB)
//   - per step (software-pipelined around the inter-WG split barrier):
//       ih GEMM (x only)  ->  barrier WAIT  ->  hh GEMM (peers' h)  ->
//       fused gates -> store h (bf16) -> barrier ARRIVE
//   - fp32 recurrent state in registers; bf16 h double-buffered through L2.
// Pre-pass: x fp32 -> bf16, t-major (removes all cvt VALU from the serial loop).

#define Bdim   64
#define Tdim   2048
#define Idim   256
#define Hdim   512
#define NWG    32
#define HC     16          // H columns per WG
#define NT     3           // N-tiles (one 16-wide tile per gate)
#define KCH    16          // K chunks of 32 for W_hh (512/32)
#define KCI    8           // K chunks of 32 for W_ih (256/32)
#define BLOCK  128         // 4 waves of 32; each wave owns 16 batch rows
#define LDS_DW ((NT*KCH + NT*KCI) * 256)        // 72 frags * 1KB = 73728 B

// workspace layout (bytes)
#define WS_CNT_OFF    0
#define WS_HBUF_OFF   1024
#define WS_HFIN_OFF   (WS_HBUF_OFF + 2*Bdim*Hdim*2)      // 132096
#define WS_XBF_OFF    (WS_HFIN_OFF + Bdim*Hdim*4)        // 263168
#define WS_ZERO_DW    (WS_XBF_OFF / 4)

typedef __attribute__((ext_vector_type(16))) __bf16 v16bf;
typedef __attribute__((ext_vector_type(8)))  float  v8f;

struct U8 { unsigned int d[8]; };

__device__ __forceinline__ unsigned short f2bf(float f) {
  unsigned u = __builtin_bit_cast(unsigned, f);
  unsigned r = (u + 0x7FFFu + ((u >> 16) & 1u)) >> 16;   // round-to-nearest-even
  return (unsigned short)r;
}

// A-fragment (16-bit, 16x32) K offset per ISA table: lanes 0-15 hold K pairs
// {0,1..6,7 | 16,17..22,23}, lanes 16-31 hold {8..15 | 24..31}.
__device__ __forceinline__ int a_koff(int j, int g) {
  return (j < 4) ? (g * 8 + 2 * j) : (16 + g * 8 + 2 * (j - 4));
}

__device__ __forceinline__ v16bf load_b_frag(const unsigned int* lds, int slot, int lane) {
  const uint4* p = (const uint4*)&lds[slot * 256 + lane * 8];
  uint4 q0 = p[0], q1 = p[1];
  U8 bu;
  bu.d[0] = q0.x; bu.d[1] = q0.y; bu.d[2] = q0.z; bu.d[3] = q0.w;
  bu.d[4] = q1.x; bu.d[5] = q1.y; bu.d[6] = q1.z; bu.d[7] = q1.w;
  return __builtin_bit_cast(v16bf, bu);
}

__global__ void gru_init_kernel(unsigned* __restrict__ ws, int n) {
  int i = blockIdx.x * blockDim.x + threadIdx.x;
  for (; i < n; i += gridDim.x * blockDim.x) ws[i] = 0u;
}

// x (B,T,I) fp32  ->  xbf (T,B,I) bf16   (one-time, HBM-bandwidth bound)
__global__ void x_pack_kernel(const float* __restrict__ x,
                              unsigned short* __restrict__ xbf) {
  const size_t total8 = (size_t)Bdim * Tdim * Idim / 8;
  size_t tid = (size_t)blockIdx.x * blockDim.x + threadIdx.x;
  for (; tid < total8; tid += (size_t)gridDim.x * blockDim.x) {
    const size_t base = tid * 8;                 // linear index in xbf (t-major)
    const int   i  = (int)(base % Idim);
    const size_t r = base / Idim;
    const int   bb = (int)(r % Bdim);
    const int   tt = (int)(r / Bdim);
    const float* src = x + ((size_t)bb * Tdim + tt) * Idim + i;
    float4 f0 = *(const float4*)src;
    float4 f1 = *(const float4*)(src + 4);
    uint4 o;
    o.x = (unsigned)f2bf(f0.x) | ((unsigned)f2bf(f0.y) << 16);
    o.y = (unsigned)f2bf(f0.z) | ((unsigned)f2bf(f0.w) << 16);
    o.z = (unsigned)f2bf(f1.x) | ((unsigned)f2bf(f1.y) << 16);
    o.w = (unsigned)f2bf(f1.z) | ((unsigned)f2bf(f1.w) << 16);
    *(uint4*)(xbf + base) = o;
  }
}

__global__ void __launch_bounds__(BLOCK, 1)
gru_persistent_kernel(const unsigned short* __restrict__ xbf,
                      const float* __restrict__ W_ih,
                      const float* __restrict__ W_hh,
                      const float* __restrict__ b,
                      const float* __restrict__ b_n,
                      const float* __restrict__ W_out,
                      const float* __restrict__ b_out,
                      float* __restrict__ out,
                      unsigned* __restrict__ ws) {
  extern __shared__ __align__(16) unsigned int lds[];

  unsigned* cnt = ws;
  unsigned short* hbuf = (unsigned short*)((char*)ws + WS_HBUF_OFF);
  float* hfinal = (float*)((char*)ws + WS_HFIN_OFF);

  const int wg   = blockIdx.x;          // 0..31 : H slice
  const int lane = threadIdx.x & 31;
  const int wid  = threadIdx.x >> 5;    // 0..3  : M tile (16 batch rows)
  const int g    = lane >> 4;
  const int mb   = wid * 16;
  const int mrow = mb + (lane & 15);

  // ---- one-time: pack this WG's weight slices into LDS as bf16 B-fragments ----
  // B-fragment layout (32K x 16N): lane L -> n = L&15, halves 2j,2j+1 -> k = a_koff(j, L>>4) + {0,1}
  for (int s = wid; s < NT * KCH + NT * KCI; s += 4) {
    const float* W; int ldk, nt, kc;
    if (s < NT * KCH) { nt = s / KCH; kc = s % KCH; W = W_hh; ldk = Hdim; }
    else { int s2 = s - NT * KCH; nt = s2 / KCI; kc = s2 % KCI; W = W_ih; ldk = Idim; }
    const int row = nt * Hdim + wg * HC + (lane & 15);   // nt == gate
    const int kb  = kc * 32;
    unsigned int* dst = &lds[s * 256 + lane * 8];
    const float* src = W + (size_t)row * ldk + kb;
#pragma unroll
    for (int j = 0; j < 8; ++j) {
      unsigned lo = f2bf(src[a_koff(j, lane >> 4)]);
      unsigned hi = f2bf(src[a_koff(j, lane >> 4) + 1]);
      dst[j] = lo | (hi << 16);
    }
  }
  __syncthreads();

  // fp32 recurrent state in registers (one column per lane, 8 m-rows)
  float hcur[8];
#pragma unroll
  for (int r = 0; r < 8; ++r) hcur[r] = 0.0f;

  const v8f vzero = {0.f, 0.f, 0.f, 0.f, 0.f, 0.f, 0.f, 0.f};

  for (int t = 0; t < Tdim; ++t) {
    const unsigned short* hprev = hbuf + (size_t)(t & 1) * Bdim * Hdim;
    unsigned short* hnext = hbuf + (size_t)((t & 1) ^ 1) * Bdim * Hdim;

    // acc[0]=r-gate, acc[1]=z-gate (ih then +hh),
    // acc[2]=candidate hh part, acc[3]=candidate ih part
    v8f acc[4];
#pragma unroll
    for (int i = 0; i < 4; ++i) acc[i] = vzero;

    // ---------- PIPELINED: ih GEMM (x only; runs before the barrier wait) ----------
    const unsigned short* xr = xbf + ((size_t)t * Bdim + mrow) * Idim;
    for (int kc = 0; kc < KCI; ++kc) {
      U8 au;
#pragma unroll
      for (int j = 0; j < 8; ++j)
        au.d[j] = *(const unsigned int*)(xr + kc * 32 + a_koff(j, g));
      v16bf A = __builtin_bit_cast(v16bf, au);
#pragma unroll
      for (int nt = 0; nt < NT; ++nt) {
        const int ai = (nt == 2) ? 3 : nt;          // candidate ih -> acc[3]
        v16bf Bf = load_b_frag(lds, NT * KCH + nt * KCI + kc, lane);
        acc[ai] = __builtin_amdgcn_wmma_f32_16x16x32_bf16(
            false, A, false, Bf, (short)0, acc[ai], false, false);
      }
    }

    // prefetch next step's x block so it is cache-resident under the barrier
    if (t + 1 < Tdim) {
      const char* nb = (const char*)(xbf + (size_t)(t + 1) * Bdim * Idim);
#pragma unroll
      for (int j = 0; j < 4; ++j)
        __builtin_prefetch(nb + ((size_t)threadIdx.x * 4 + j) * 128, 0, 0);
    }

    // ---------- barrier WAIT: peers' h for step t ready ----------
    if (threadIdx.x == 0 && t > 0) {
      const unsigned tgt = (unsigned)t * NWG;
      while (*(volatile unsigned*)cnt < tgt) __builtin_amdgcn_s_sleep(1);
    }
    __syncthreads();
    __threadfence();        // acquire peers' h stores

    // ---------- hh GEMM: h(bf16, 64x512) x W_hh^T slice (512x48) ----------
    for (int kc = 0; kc < KCH; ++kc) {
      U8 au;
      const unsigned short* hp = hprev + (size_t)mrow * Hdim + kc * 32;
#pragma unroll
      for (int j = 0; j < 8; ++j)
        au.d[j] = *(const unsigned int*)(hp + a_koff(j, g));
      v16bf A = __builtin_bit_cast(v16bf, au);
#pragma unroll
      for (int nt = 0; nt < NT; ++nt) {
        v16bf Bf = load_b_frag(lds, nt * KCH + kc, lane);
        acc[nt] = __builtin_amdgcn_wmma_f32_16x16x32_bf16(
            false, A, false, Bf, (short)0, acc[nt], false, false);
      }
    }

    // ---------- fused gates (C/D layout: lane->n=L&15, vgpr r -> m = r + g*8) ----------
    {
      const int col = wg * HC + (lane & 15);
      const float br = b[col];
      const float bz = b[Hdim + col];
      const float bg = b[2 * Hdim + col];
      const float bn = b_n[col];
#pragma unroll
      for (int r = 0; r < 8; ++r) {
        const float ar  = acc[0][r] + br;
        const float az  = acc[1][r] + bz;
        const float agh = acc[2][r];               // hh candidate part
        const float agi = acc[3][r] + bg;          // ih candidate part (+bias)
        const float rv = __builtin_amdgcn_rcpf(1.0f + __expf(-ar));
        const float zv = __builtin_amdgcn_rcpf(1.0f + __expf(-az));
        const float ta = agi + rv * (agh + bn);
        const float gv = 1.0f - 2.0f * __builtin_amdgcn_rcpf(__expf(2.0f * ta) + 1.0f);
        const float hn = gv + zv * (hcur[r] - gv);
        hcur[r] = hn;
        const int m = mb + g * 8 + r;
        hnext[(size_t)m * Hdim + col] = f2bf(hn);
        if (t == Tdim - 1) hfinal[(size_t)m * Hdim + col] = hn;
      }
    }

    // ---------- barrier ARRIVE (no wait here; next step waits before hh GEMM) ----
    __threadfence();        // release h stores
    __syncthreads();
    if (threadIdx.x == 0) atomicAdd(cnt, 1u);
  }

  // ---------- final projection: out = h_final @ W_out^T + b_out (WG0 only) ----------
  if (wg == 0) {
    if (threadIdx.x == 0) {
      const unsigned tgt = (unsigned)Tdim * NWG;
      while (*(volatile unsigned*)cnt < tgt) __builtin_amdgcn_s_sleep(1);
    }
    __syncthreads();
    __threadfence();
    if (threadIdx.x < Bdim) {
      const int bi = threadIdx.x;
      float s = 0.0f;
      for (int k = 0; k < Hdim; ++k)
        s += hfinal[(size_t)bi * Hdim + k] * W_out[k];
      out[bi] = s + b_out[0];
    }
  }
}

extern "C" void kernel_launch(void* const* d_in, const int* in_sizes, int n_in,
                              void* d_out, int out_size, void* d_ws, size_t ws_size,
                              hipStream_t stream) {
  (void)in_sizes; (void)n_in; (void)out_size; (void)ws_size;
  const float* x     = (const float*)d_in[0];
  const float* W_ih  = (const float*)d_in[1];
  const float* W_hh  = (const float*)d_in[2];
  const float* b     = (const float*)d_in[3];
  const float* b_n   = (const float*)d_in[4];
  const float* W_out = (const float*)d_in[5];
  const float* b_out = (const float*)d_in[6];
  float* out = (float*)d_out;
  unsigned* ws = (unsigned*)d_ws;
  unsigned short* xbf = (unsigned short*)((char*)d_ws + WS_XBF_OFF);

  // zero barrier counter + h double buffer + hfinal (deterministic per launch)
  gru_init_kernel<<<64, 256, 0, stream>>>(ws, WS_ZERO_DW);

  // one-time x fp32 -> bf16 (t-major) pre-pass
  x_pack_kernel<<<8192, 256, 0, stream>>>(x, xbf);

  gru_persistent_kernel<<<NWG, BLOCK, LDS_DW * sizeof(unsigned int), stream>>>(
      xbf, W_ih, W_hh, b, b_n, W_out, b_out, out, ws);
}
